// DiffusionGraphConv_71983651881364
// MI455X (gfx1250) — matrix operand order repaired and checked by reading, not compiled, not run on hardware
//
#include <hip/hip_runtime.h>
#include <hip/hip_bf16.h>
#include <stdint.h>

// ---------------------------------------------------------------------------
// DCRNN diffusion graph conv for MI455X (gfx1250, wave32)
// bf16 WMMA + double-buffered async global->LDS tile staging (ASYNCcnt path)
// ---------------------------------------------------------------------------

typedef __attribute__((ext_vector_type(16))) __bf16 v16bf;
typedef __attribute__((ext_vector_type(8)))  __bf16 v8bf;
typedef __attribute__((ext_vector_type(8)))  float  v8f;
typedef __attribute__((ext_vector_type(4)))  float  v4f;
typedef int v4i_g __attribute__((vector_size(16)));   // GCC vector, matches builtin proto

#define NNODES 1024
#define NBATCH 64
#define DMODEL 128            // input_size = 64 + 64
#define OUTDIM 128
#define BN     (NBATCH * NNODES)

// padded LDS strides (elements); multiple of 8 keeps 16B alignment for v8bf loads
#define LDA 40
#define LDB 40

union FragAB { v16bf v; v8bf h[2]; };
struct alignas(8)  bf16x4 { __bf16 a, b, c, d; };
struct alignas(16) bf16x8 { __bf16 v[8]; };

__device__ __forceinline__ v8f wmma_bf16(const v16bf& a, const v16bf& b, const v8f& c) {
  return __builtin_amdgcn_wmma_f32_16x16x32_bf16(false, a, false, b, (short)0, c, false, false);
}

// ---- gfx1250 async global -> LDS copy (16B per lane), ASYNCcnt-tracked -----
__device__ __forceinline__ void async_b128(const void* g, void* lds) {
#if __has_builtin(__builtin_amdgcn_global_load_async_to_lds_b128)
  __builtin_amdgcn_global_load_async_to_lds_b128(
      (__attribute__((address_space(1))) v4i_g*)g,
      (__attribute__((address_space(3))) v4i_g*)lds, 0, 0);
#else
  unsigned loff = (unsigned)(uintptr_t)lds;   // low 32 bits of generic = LDS offset
  asm volatile("global_load_async_to_lds_b128 %0, %1, off"
               :: "v"(loff), "v"(g) : "memory");
#endif
}

__device__ __forceinline__ void wait_async0() {
#if __has_builtin(__builtin_amdgcn_s_wait_asynccnt)
  __builtin_amdgcn_s_wait_asynccnt(0);
#else
  asm volatile("s_wait_asynccnt 0x0" ::: "memory");
#endif
}

// ---------------------------------------------------------------------------
// supports f32 -> bf16 (row-major copy)
// ---------------------------------------------------------------------------
__global__ void cvt_support_kernel(const float* __restrict__ s, __bf16* __restrict__ d) {
  int gid = blockIdx.x * blockDim.x + threadIdx.x;     // one float4
  v4f v = *(const v4f*)(s + (long)gid * 4);
  bf16x4 o = {(__bf16)v.x, (__bf16)v.y, (__bf16)v.z, (__bf16)v.w};
  *(bf16x4*)(d + (long)gid * 4) = o;
}

// ---------------------------------------------------------------------------
// X0 = concat(inputs, state):  f32 [B,N,128]  +  bf16 transposed [B,128,N]
// ---------------------------------------------------------------------------
__global__ void build_x0_kernel(const float* __restrict__ in,
                                const float* __restrict__ st,
                                float* __restrict__ x0,
                                __bf16* __restrict__ x0t) {
  int gid = blockIdx.x * blockDim.x + threadIdx.x;     // (b, d4, n)
  int b   = gid >> 15;
  int rem = gid & 32767;
  int d   = (rem >> 10) * 4;
  int n   = rem & 1023;
  const float* src = (d < 64) ? (in + (long)b * 65536 + n * 64 + d)
                              : (st + (long)b * 65536 + n * 64 + (d - 64));
  v4f v = *(const v4f*)src;
  *(v4f*)(x0 + (long)b * (NNODES * DMODEL) + n * DMODEL + d) = v;
  __bf16* t = x0t + (long)b * (DMODEL * NNODES) + d * NNODES + n;
  t[0]        = (__bf16)v.x;
  t[NNODES]   = (__bf16)v.y;
  t[2*NNODES] = (__bf16)v.z;
  t[3*NNODES] = (__bf16)v.w;
}

// ---------------------------------------------------------------------------
// Y[b]  = S @ X[b]            (mode 0)   [+ bf16 transposed copy YT if given]
// Y[b]  = 2*(S @ X[b]) - P[b] (mode 1)
// S: bf16 [N,N], XT: bf16 [B,128,N] (transposed!), Y/P: f32 [B,N,128]
// grid = (N/128, B), block = 256 (8 waves). Double-buffered async staging.
// ---------------------------------------------------------------------------
__global__ __launch_bounds__(256)
void diffusion_spmm_kernel(const __bf16* __restrict__ S,
                           const __bf16* __restrict__ XT,
                           float* __restrict__ Y,
                           __bf16* __restrict__ YT,
                           const float* __restrict__ P,
                           int mode) {
  __shared__ __align__(16) __bf16 lA[2][128 * LDA];  // A tile: [i][k]
  __shared__ __align__(16) __bf16 lB[2][128 * LDB];  // B tile: [d][k]

  const int tid  = threadIdx.x;
  const int b    = blockIdx.y;
  const int i0   = blockIdx.x * 128;

  const int w    = tid >> 5;
  const int lane = tid & 31;
  const int wr   = w & 3;             // 32-row strip
  const int wc   = w >> 2;            // 64-col half
  const int half = lane >> 4;
  const int l16  = lane & 15;

  const __bf16* XTb = XT + (long)b * (DMODEL * NNODES);

  // Per-thread copy job: both tiles are 128 rows x 64B; each thread moves a
  // 32B chunk of one row of each tile (2 x b128 per tile).
  const int cp_row = tid >> 1;                 // 0..127
  const int cp_sg  = (tid & 1) * 2;            // 16B segment base: 0 or 2
  const __bf16* Srow = S   + (long)(i0 + cp_row) * NNODES + cp_sg * 8;
  const __bf16* Xrow = XTb + (long)cp_row      * NNODES + cp_sg * 8;
  const int loA = cp_row * LDA + cp_sg * 8;
  const int loB = cp_row * LDB + cp_sg * 8;

  v8f acc[2][4] = {};

  constexpr int KT = NNODES / 32;

  // prologue: stage tile 0 into buffer 0
  async_b128(Srow,     &lA[0][loA]);
  async_b128(Srow + 8, &lA[0][loA + 8]);
  async_b128(Xrow,     &lB[0][loB]);
  async_b128(Xrow + 8, &lB[0][loB + 8]);

  for (int kc = 0; kc < KT; ++kc) {
    const int cur = kc & 1;
    const int nxt = cur ^ 1;

    wait_async0();        // this wave's copies for tile kc are done
    __syncthreads();      // all waves' copies done; compute kc-1 finished

    if (kc + 1 < KT) {    // stage tile kc+1 into the other buffer (overlaps WMMA)
      const int k1 = (kc + 1) * 32;
      async_b128(Srow + k1,     &lA[nxt][loA]);
      async_b128(Srow + k1 + 8, &lA[nxt][loA + 8]);
      async_b128(Xrow + k1,     &lB[nxt][loB]);
      async_b128(Xrow + k1 + 8, &lB[nxt][loB + 8]);
    }

    // ---- fragments per documented 16-bit WMMA layouts (wave32)
    FragAB afr[2], bfr[4];
    const int akb = half ? 8 : 0;     // A: lanes 0-15 K{0-7,16-23}; 16-31 K{8-15,24-31}
    const int bkb = half ? 8 : 0;     // B: lanes 0-15 K{0-7,16-23}; 16-31 K{8-15,24-31}
    #pragma unroll
    for (int mt = 0; mt < 2; ++mt) {
      int m = 32 * wr + 16 * mt + l16;
      afr[mt].h[0] = *(const v8bf*)(&lA[cur][m * LDA + akb]);
      afr[mt].h[1] = *(const v8bf*)(&lA[cur][m * LDA + akb + 16]);
    }
    #pragma unroll
    for (int nt = 0; nt < 4; ++nt) {
      int n = 64 * wc + 16 * nt + l16;
      bfr[nt].h[0] = *(const v8bf*)(&lB[cur][n * LDB + bkb]);
      bfr[nt].h[1] = *(const v8bf*)(&lB[cur][n * LDB + bkb + 16]);
    }
    #pragma unroll
    for (int mt = 0; mt < 2; ++mt)
      #pragma unroll
      for (int nt = 0; nt < 4; ++nt)
        acc[mt][nt] = wmma_bf16(afr[mt].v, bfr[nt].v, acc[mt][nt]);
  }

  // ---- epilogue: C layout VGPR j -> M = j + 8*half, N = l16
  float* Yb = Y + (long)b * (NNODES * DMODEL);
  const float* Pb = P ? (P + (long)b * (NNODES * DMODEL)) : nullptr;
  __bf16* YTb = YT ? (YT + (long)b * (DMODEL * NNODES)) : nullptr;
  #pragma unroll
  for (int mt = 0; mt < 2; ++mt) {
    #pragma unroll
    for (int nt = 0; nt < 4; ++nt) {
      int n = 64 * wc + 16 * nt + l16;
      #pragma unroll
      for (int j = 0; j < 8; ++j) {
        int m = 32 * wr + 16 * mt + j + half * 8;
        long idx = (long)(i0 + m) * DMODEL + n;
        float v = acc[mt][nt][j];
        if (mode == 1) v = 2.0f * v - Pb[idx];
        Yb[idx] = v;
      }
      if (YTb) {
        // 8 consecutive node rows per lane -> one 16B transposed bf16 store
        int node0 = i0 + 32 * wr + 16 * mt + half * 8;
        bf16x8 pk;
        #pragma unroll
        for (int j = 0; j < 8; ++j) pk.v[j] = (__bf16)acc[mt][nt][j];
        *(bf16x8*)(YTb + (long)n * NNODES + node0) = pk;
      }
    }
  }
}

// ---------------------------------------------------------------------------
// out[r, o] = sum_{m,d} XS[m][r][d] * W[m*128+d][o] + bias[o]
// XS: [5][BN][128] f32, W: [640][128] f32.  grid = BN/128, block = 256
// ---------------------------------------------------------------------------
__global__ __launch_bounds__(256)
void proj_gemm_kernel(const float* __restrict__ XS,
                      const float* __restrict__ W,
                      const float* __restrict__ bias,
                      float* __restrict__ out) {
  __shared__ __align__(16) __bf16 lA[128 * LDA];
  __shared__ __align__(16) __bf16 lB[128 * LDB];

  const int tid  = threadIdx.x;
  const long r0  = (long)blockIdx.x * 128;

  const int w    = tid >> 5;
  const int lane = tid & 31;
  const int wr   = w & 3;
  const int wc   = w >> 2;
  const int half = lane >> 4;
  const int l16  = lane & 15;

  const int a_c4 = tid & 7;
  const int a_r  = tid >> 3;
  const int b_c4 = tid & 31;
  const int b_k  = tid >> 5;

  v8f acc[2][4] = {};

  for (int kc = 0; kc < 640 / 32; ++kc) {
    const int m  = kc >> 2;            // diffusion matrix index
    const int d0 = (kc & 3) * 32;      // feature column base within that matrix
    const float* A = XS + (long)m * ((long)BN * DMODEL);

    #pragma unroll
    for (int p = 0; p < 4; ++p) {
      int row = a_r + 32 * p;
      const float* gp = A + (r0 + row) * DMODEL + d0 + a_c4 * 4;
      v4f v = *(const v4f*)gp;
      if (kc + 1 < 20) __builtin_prefetch(gp + 32, 0, 1);
      bf16x4 o = {(__bf16)v.x, (__bf16)v.y, (__bf16)v.z, (__bf16)v.w};
      *(bf16x4*)(lA + row * LDA + a_c4 * 4) = o;
    }
    #pragma unroll
    for (int p = 0; p < 4; ++p) {
      int k = b_k + 8 * p;
      v4f v = *(const v4f*)(W + (long)(kc * 32 + k) * OUTDIM + b_c4 * 4);
      lB[(b_c4 * 4 + 0) * LDB + k] = (__bf16)v.x;
      lB[(b_c4 * 4 + 1) * LDB + k] = (__bf16)v.y;
      lB[(b_c4 * 4 + 2) * LDB + k] = (__bf16)v.z;
      lB[(b_c4 * 4 + 3) * LDB + k] = (__bf16)v.w;
    }
    __syncthreads();

    FragAB afr[2], bfr[4];
    const int akb = half ? 8 : 0;
    const int bkb = half ? 8 : 0;
    #pragma unroll
    for (int mt = 0; mt < 2; ++mt) {
      int mm = 32 * wr + 16 * mt + l16;
      afr[mt].h[0] = *(const v8bf*)(lA + mm * LDA + akb);
      afr[mt].h[1] = *(const v8bf*)(lA + mm * LDA + akb + 16);
    }
    #pragma unroll
    for (int nt = 0; nt < 4; ++nt) {
      int n = 64 * wc + 16 * nt + l16;
      bfr[nt].h[0] = *(const v8bf*)(lB + n * LDB + bkb);
      bfr[nt].h[1] = *(const v8bf*)(lB + n * LDB + bkb + 16);
    }
    #pragma unroll
    for (int mt = 0; mt < 2; ++mt)
      #pragma unroll
      for (int nt = 0; nt < 4; ++nt)
        acc[mt][nt] = wmma_bf16(afr[mt].v, bfr[nt].v, acc[mt][nt]);

    __syncthreads();
  }

  #pragma unroll
  for (int mt = 0; mt < 2; ++mt) {
    #pragma unroll
    for (int nt = 0; nt < 4; ++nt) {
      int n = 64 * wc + 16 * nt + l16;
      float bv = bias[n];
      #pragma unroll
      for (int j = 0; j < 8; ++j) {
        int mm = 32 * wr + 16 * mt + j + half * 8;
        out[(r0 + mm) * OUTDIM + n] = acc[mt][nt][j] + bv;
      }
    }
  }
}

// ---------------------------------------------------------------------------
extern "C" void kernel_launch(void* const* d_in, const int* in_sizes, int n_in,
                              void* d_out, int out_size, void* d_ws, size_t ws_size,
                              hipStream_t stream) {
  (void)in_sizes; (void)n_in; (void)out_size; (void)ws_size;
  const float* supports = (const float*)d_in[0];   // [2, 1024, 1024]
  const float* inputs   = (const float*)d_in[1];   // [64, 1024*64]
  const float* state    = (const float*)d_in[2];   // [64, 1024*64]
  const float* weight   = (const float*)d_in[3];   // [640, 128]
  const float* biases   = (const float*)d_in[4];   // [128]
  float* out = (float*)d_out;                      // [64, 1024*128]

  const long MSZ = (long)BN * DMODEL;              // 8,388,608 elements
  float*  XS  = (float*)d_ws;                      // f32 [5][BN][128]  (160 MB)
  __bf16* Sbf = (__bf16*)(XS + 5 * MSZ);           // bf16 [2][N][N]    (4 MB)
  __bf16* X0T = Sbf + 2L * NNODES * NNODES;        // bf16 [B][128][N]  (16 MB)
  __bf16* XAT = X0T + MSZ;
  __bf16* XCT = XAT + MSZ;

  float* X0 = XS;
  float* XA = XS + 1 * MSZ;
  float* XB = XS + 2 * MSZ;
  float* XC = XS + 3 * MSZ;
  float* XE = XS + 4 * MSZ;
  const __bf16* S0 = Sbf;
  const __bf16* S1 = Sbf + (long)NNODES * NNODES;

  cvt_support_kernel<<<(2L * NNODES * NNODES / 4) / 256, 256, 0, stream>>>(supports, Sbf);
  build_x0_kernel<<<(BN * DMODEL / 4) / 256, 256, 0, stream>>>(inputs, state, X0, X0T);

  dim3 g(NNODES / 128, NBATCH);
  diffusion_spmm_kernel<<<g, 256, 0, stream>>>(S0, X0T, XA, XAT, nullptr, 0); // XA = S0 X0
  diffusion_spmm_kernel<<<g, 256, 0, stream>>>(S0, XAT, XB, nullptr, X0, 1);  // XB = 2 S0 XA - X0
  diffusion_spmm_kernel<<<g, 256, 0, stream>>>(S1, XAT, XC, XCT, nullptr, 0); // XC = S1 XA
  diffusion_spmm_kernel<<<g, 256, 0, stream>>>(S1, XCT, XE, nullptr, XA, 1);  // XE = 2 S1 XC - XA

  proj_gemm_kernel<<<BN / 128, 256, 0, stream>>>(XS, weight, biases, out);
}
